// KGAT_61040075210791
// MI455X (gfx1250) — matrix operand order repaired and checked by reading, not compiled, not run on hardware
//
#include <hip/hip_runtime.h>
#include <math.h>

typedef __attribute__((ext_vector_type(16))) _Float16 v16h;
typedef __attribute__((ext_vector_type(8)))  float    v8f;

#define NTOT   100000
#define NUSERS 30000
#define DDIM   64

// ---------------------------------------------------------------------------
// ego0 = concat(user_tab, entity_tab); also writes out[:, 0:64]
// ---------------------------------------------------------------------------
__global__ void concat_init(const float* __restrict__ u, const float* __restrict__ e,
                            float* __restrict__ ego0, float* __restrict__ out) {
    int idx = blockIdx.x * blockDim.x + threadIdx.x;
    if (idx >= NTOT * DDIM) return;
    int row = idx >> 6, col = idx & 63;
    float v = (row < NUSERS) ? u[idx] : e[idx - NUSERS * DDIM];
    ego0[idx] = v;
    out[(long)row * 160 + col] = v;
}

// ---------------------------------------------------------------------------
// One-time weight prep: split f32 W into (hi, lo) f16 pair, stored pre-swizzled
// in the exact 32x16 WMMA B-fragment order:
//   off = ((((nt*2 + s)*2 + h)*16 + col)*16) + i
// so a lane's whole fragment is 32 contiguous bytes (two b128 loads).
// ---------------------------------------------------------------------------
template<int DOUT>
__global__ void split_weights(const float* __restrict__ W1, const float* __restrict__ W2,
                              _Float16* __restrict__ W1h, _Float16* __restrict__ W1l,
                              _Float16* __restrict__ W2h, _Float16* __restrict__ W2l) {
    int tid = blockIdx.x * blockDim.x + threadIdx.x;
    if (tid >= 64 * DOUT) return;
    int k = tid / DOUT, n = tid % DOUT;
    int nt = n >> 4, c = n & 15;
    int s = k >> 5, kk = k & 31, hh = (kk >> 3) & 1, i = (kk & 7) + ((kk >> 4) << 3);
    int off = ((((nt * 2 + s) * 2 + hh) * 16 + c) * 16) + i;
    float w1 = W1[tid], w2 = W2[tid];
    _Float16 h1 = (_Float16)w1, h2 = (_Float16)w2;
    W1h[off] = h1; W1l[off] = (_Float16)(w1 - (float)h1);
    W2h[off] = h2; W2l[off] = (_Float16)(w2 - (float)h2);
}

// ---------------------------------------------------------------------------
// COO SpMM scatter: y[rows[e],:] += vals[e] * x[cols[e],:]
// 16 lanes per edge, float4 gather (node table is L2-resident: 25.6MB << 192MB),
// 4 f32 atomics/lane. Edge streams are the only HBM traffic -> prefetch ahead.
// ---------------------------------------------------------------------------
__global__ void spmm_scatter(const int* __restrict__ rows, const int* __restrict__ cols,
                             const float* __restrict__ vals, const float* __restrict__ x,
                             float* __restrict__ y, int nedges) {
    long tid = (long)blockIdx.x * blockDim.x + threadIdx.x;
    long e   = tid >> 4;
    if (e >= nedges) return;
    int   q = (int)(tid & 15);
    if (q == 0 && e + 4096 < nedges) {          // global_prefetch_b8 on edge streams
        __builtin_prefetch(rows + e + 4096, 0, 1);
        __builtin_prefetch(cols + e + 4096, 0, 1);
        __builtin_prefetch(vals + e + 4096, 0, 1);
    }
    int   r = rows[e], c = cols[e];
    float v = vals[e];
    const float4 m = *(const float4*)(x + (long)c * DDIM + q * 4);
    float* dst = y + (long)r * DDIM + q * 4;
    atomicAdd(dst + 0, m.x * v);
    atomicAdd(dst + 1, m.y * v);
    atomicAdd(dst + 2, m.z * v);
    atomicAdd(dst + 3, m.w * v);
}

// ---------------------------------------------------------------------------
// Fused bi-interaction layer on the WMMA pipe.
//   sum_e = lrelu((ego+side) @ W1 + b1)
//   bi_e  = lrelu((ego*side) @ W2 + b2)
//   out[:, COLBASE:COLBASE+DOUT] = l2norm(sum_e + bi_e)   (row-wise)
// One wave32 per 16-row tile. fp32 GEMM emulated as hi/lo fp16 split:
//   A*B ~= Ahi*Bhi + Alo*Bhi + Ahi*Blo   (3 v_wmma per tile, ~2^-22 rel err)
// B fragments are pre-split/pre-swizzled -> pure vector loads here.
// ---------------------------------------------------------------------------
template<int DOUT, int COLBASE, bool WRITE_EGO>
__launch_bounds__(256)
__global__ void bi_layer(const float* __restrict__ ego, const float* __restrict__ side,
                         const _Float16* __restrict__ W1h, const _Float16* __restrict__ W1l,
                         const _Float16* __restrict__ W2h, const _Float16* __restrict__ W2l,
                         const float* __restrict__ b1, const float* __restrict__ b2,
                         float* __restrict__ out, float* __restrict__ ego_next) {
    constexpr int NT = DOUT / 16;
    const int wave = threadIdx.x >> 5;
    const int tile = blockIdx.x * 8 + wave;
    const int r0   = tile * 16;
    if (r0 >= NTOT) return;            // wave-uniform: EXEC stays all-ones below
    const int l = threadIdx.x & 31;
    const int m = l & 15;              // A-row within tile / B,C column index
    const int h = l >> 4;              // half-wave selector

    const float* erow = ego  + (long)(r0 + m) * DDIM;
    const float* srow = side + (long)(r0 + m) * DDIM;

    // Build A fragments (hi/lo split) for both branches, both 32-wide K steps.
    // 16-bit A 16x32 layout: element i of the lane's v16h holds
    //   k = 32*s + (i&7) + 8*((i>>3)*2 + h)
    v16h as_hi[2], as_lo[2], ab_hi[2], ab_lo[2];
#pragma unroll
    for (int s = 0; s < 2; ++s) {
        const int k0 = 32 * s + 8 * h;        // elements 0..7
        const int k1 = 32 * s + 16 + 8 * h;   // elements 8..15
        float4 ea = *(const float4*)(erow + k0);
        float4 eb = *(const float4*)(erow + k0 + 4);
        float4 ec = *(const float4*)(erow + k1);
        float4 ed = *(const float4*)(erow + k1 + 4);
        float4 sa = *(const float4*)(srow + k0);
        float4 sb = *(const float4*)(srow + k0 + 4);
        float4 sc = *(const float4*)(srow + k1);
        float4 sd = *(const float4*)(srow + k1 + 4);
        const float ef[16] = {ea.x, ea.y, ea.z, ea.w, eb.x, eb.y, eb.z, eb.w,
                              ec.x, ec.y, ec.z, ec.w, ed.x, ed.y, ed.z, ed.w};
        const float sf[16] = {sa.x, sa.y, sa.z, sa.w, sb.x, sb.y, sb.z, sb.w,
                              sc.x, sc.y, sc.z, sc.w, sd.x, sd.y, sd.z, sd.w};
#pragma unroll
        for (int i = 0; i < 16; ++i) {
            float su = ef[i] + sf[i];
            float pr = ef[i] * sf[i];
            _Float16 sh = (_Float16)su;
            _Float16 ph = (_Float16)pr;
            as_hi[s][i] = sh; as_lo[s][i] = (_Float16)(su - (float)sh);
            ab_hi[s][i] = ph; ab_lo[s][i] = (_Float16)(pr - (float)ph);
        }
    }

    float vals_[NT][8];
    float ss[8];
#pragma unroll
    for (int r = 0; r < 8; ++r) ss[r] = 0.0f;

#pragma unroll
    for (int nt = 0; nt < NT; ++nt) {
        v8f accS = {}; v8f accB = {};
        const int n = nt * 16 + m;            // output column this lane owns
#pragma unroll
        for (int s = 0; s < 2; ++s) {
            // Pre-swizzled B fragments: 32 contiguous bytes per lane.
            const long foff = (long)((((nt * 2 + s) * 2 + h) * 16 + m) * 16);
            v16h f1h = *(const v16h*)(W1h + foff);
            v16h f1l = *(const v16h*)(W1l + foff);
            v16h f2h = *(const v16h*)(W2h + foff);
            v16h f2l = *(const v16h*)(W2l + foff);
            accS = __builtin_amdgcn_wmma_f32_16x16x32_f16(false, as_hi[s], false, f1h, (short)0, accS, false, false);
            accS = __builtin_amdgcn_wmma_f32_16x16x32_f16(false, as_lo[s], false, f1h, (short)0, accS, false, false);
            accS = __builtin_amdgcn_wmma_f32_16x16x32_f16(false, as_hi[s], false, f1l, (short)0, accS, false, false);
            accB = __builtin_amdgcn_wmma_f32_16x16x32_f16(false, ab_hi[s], false, f2h, (short)0, accB, false, false);
            accB = __builtin_amdgcn_wmma_f32_16x16x32_f16(false, ab_lo[s], false, f2h, (short)0, accB, false, false);
            accB = __builtin_amdgcn_wmma_f32_16x16x32_f16(false, ab_hi[s], false, f2l, (short)0, accB, false, false);
        }
        const float bb1 = b1[n];
        const float bb2 = b2[n];
#pragma unroll
        for (int r = 0; r < 8; ++r) {         // C/D: VGPR r -> row M = r + 8*h
            float sv = accS[r] + bb1; sv = sv > 0.0f ? sv : 0.01f * sv;
            float bv = accB[r] + bb2; bv = bv > 0.0f ? bv : 0.01f * bv;
            float v  = sv + bv;
            vals_[nt][r] = v;
            ss[r] += v * v;
        }
    }

    // Row-wise L2 norm: reduce over the 16 lanes of this half-wave.
#pragma unroll
    for (int r = 0; r < 8; ++r) {
        float s = ss[r];
        s += __shfl_xor(s, 1, 16);
        s += __shfl_xor(s, 2, 16);
        s += __shfl_xor(s, 4, 16);
        s += __shfl_xor(s, 8, 16);
        ss[r] = 1.0f / fmaxf(sqrtf(s), 1e-12f);
    }

#pragma unroll
    for (int nt = 0; nt < NT; ++nt) {
#pragma unroll
        for (int r = 0; r < 8; ++r) {
            const int grow = r0 + r + 8 * h;
            const int col  = nt * 16 + m;
            float v = vals_[nt][r] * ss[r];
            out[(long)grow * 160 + COLBASE + col] = v;
            if (WRITE_EGO) ego_next[(long)grow * DDIM + col] = v;
        }
    }
}

// ---------------------------------------------------------------------------
extern "C" void kernel_launch(void* const* d_in, const int* in_sizes, int n_in,
                              void* d_out, int out_size, void* d_ws, size_t ws_size,
                              hipStream_t stream) {
    const float* user_tab   = (const float*)d_in[0];
    const float* entity_tab = (const float*)d_in[1];
    const float* A_vals     = (const float*)d_in[2];
    const float* W1_1 = (const float*)d_in[3];
    const float* b1_1 = (const float*)d_in[4];
    const float* W2_1 = (const float*)d_in[5];
    const float* b2_1 = (const float*)d_in[6];
    const float* W1_2 = (const float*)d_in[7];
    const float* b1_2 = (const float*)d_in[8];
    const float* W2_2 = (const float*)d_in[9];
    const float* b2_2 = (const float*)d_in[10];
    const int* A_rows = (const int*)d_in[11];
    const int* A_cols = (const int*)d_in[12];
    float* out = (float*)d_out;
    const int E = in_sizes[2];

    // Workspace carve-out: 3 x [N,64] f32 (76.8 MB) + split-weight buffers (48 KB)
    float* ego0 = (float*)d_ws;
    float* side = ego0 + (size_t)NTOT * DDIM;
    float* ego1 = side + (size_t)NTOT * DDIM;
    _Float16* wb = (_Float16*)(ego1 + (size_t)NTOT * DDIM);
    _Float16* W1h_1 = wb;            _Float16* W1l_1 = wb + 4096;
    _Float16* W2h_1 = wb + 2 * 4096; _Float16* W2l_1 = wb + 3 * 4096;
    _Float16* wb2 = wb + 4 * 4096;
    _Float16* W1h_2 = wb2;            _Float16* W1l_2 = wb2 + 2048;
    _Float16* W2h_2 = wb2 + 2 * 2048; _Float16* W2l_2 = wb2 + 3 * 2048;

    // Weight prep (tiny, once per call)
    split_weights<64><<<(64 * 64 + 255) / 256, 256, 0, stream>>>(
        W1_1, W2_1, W1h_1, W1l_1, W2h_1, W2l_1);
    split_weights<32><<<(64 * 32 + 255) / 256, 256, 0, stream>>>(
        W1_2, W2_2, W1h_2, W1l_2, W2h_2, W2l_2);

    const int total = NTOT * DDIM;
    concat_init<<<(total + 255) / 256, 256, 0, stream>>>(user_tab, entity_tab, ego0, out);

    const long spmm_threads = (long)E * 16;
    const int  spmm_blocks  = (int)((spmm_threads + 255) / 256);
    const int  tiles        = NTOT / 16;            // 6250, exact
    const int  gemm_blocks  = (tiles + 7) / 8;      // 8 waves / block

    // Layer 1: 64 -> 64
    hipMemsetAsync(side, 0, (size_t)NTOT * DDIM * sizeof(float), stream);
    spmm_scatter<<<spmm_blocks, 256, 0, stream>>>(A_rows, A_cols, A_vals, ego0, side, E);
    bi_layer<64, 64, true><<<gemm_blocks, 256, 0, stream>>>(
        ego0, side, W1h_1, W1l_1, W2h_1, W2l_1, b1_1, b2_1, out, ego1);

    // Layer 2: 64 -> 32
    hipMemsetAsync(side, 0, (size_t)NTOT * DDIM * sizeof(float), stream);
    spmm_scatter<<<spmm_blocks, 256, 0, stream>>>(A_rows, A_cols, A_vals, ego1, side, E);
    bi_layer<32, 128, false><<<gemm_blocks, 256, 0, stream>>>(
        ego1, side, W1h_2, W1l_2, W2h_2, W2l_2, b1_2, b2_2, out, nullptr);
}
